// EditDistStatModel_14328010899516
// MI455X (gfx1250) — matrix operand order repaired and checked by reading, not compiled, not run on hardware
//
#include <hip/hip_runtime.h>
#include <math.h>

#define TT 1024
#define VV 1024
#define AA 256
#define EE 256
#define CC (AA + EE + AA * EE) /* 66048 */
#define NEGINF (-__builtin_inff())

typedef __attribute__((ext_vector_type(16))) _Float16    v16h;
typedef __attribute__((ext_vector_type(8)))  float       v8f;
typedef __attribute__((ext_vector_type(4)))  unsigned int v4u;
typedef __attribute__((ext_vector_type(8)))  int         v8i;
typedef __attribute__((ext_vector_type(4)))  int         v4i;

// ---------------- numerics helpers ----------------
__device__ __forceinline__ float lse2(float a, float b) {
    float mx = fmaxf(a, b), mn = fminf(a, b);
    if (mx == NEGINF) return NEGINF;
    return mx + log1pf(__expf(mn - mx));
}

__device__ __forceinline__ float wave_incl(float v, int lane) {
#pragma unroll
    for (int d = 1; d < 32; d <<= 1) {
        float o = __shfl_up(v, d, 32);
        if (lane >= d) v = lse2(v, o);
    }
    return v;
}
__device__ __forceinline__ float wave_suff(float v, int lane) {
#pragma unroll
    for (int d = 1; d < 32; d <<= 1) {
        float o = __shfl_down(v, d, 32);
        if (lane < 32 - d) v = lse2(v, o);
    }
    return v;
}

// ordered-uint encoding of float for atomicMax (monotonic incl. -inf)
__device__ __forceinline__ unsigned int fkey(float f) {
    unsigned int b = __float_as_uint(f);
    return (b & 0x80000000u) ? ~b : (b | 0x80000000u);
}
__device__ __forceinline__ float fdec(unsigned int k) {
    unsigned int b = (k & 0x80000000u) ? (k & 0x7fffffffu) : ~k;
    return __uint_as_float(b);
}

// ---------------- TDM staging of one 256-float weight row into LDS ----------
#if __has_builtin(__builtin_amdgcn_tensor_load_to_lds) && __has_builtin(__builtin_amdgcn_s_wait_tensorcnt)
#define HAVE_TDM 1
#else
#define HAVE_TDM 0
#endif

__device__ __forceinline__ void stage_row(const float* gsrc, float* lds_dst, int lane) {
#if HAVE_TDM
    unsigned long long ga = (unsigned long long)(uintptr_t)gsrc;
    unsigned int ldsoff   = (unsigned int)(uintptr_t)lds_dst; // low 32 bits = LDS offset
    v4u g0;
    g0[0] = 1u;                                         // count=1, user descriptor
    g0[1] = ldsoff;                                     // lds_addr
    g0[2] = (unsigned int)(ga & 0xffffffffu);           // global_addr lo
    g0[3] = (unsigned int)((ga >> 32) & 0x01ffffffu) | (2u << 30); // addr hi | type=2
    v8i g1;
    g1[0] = (int)(2u << 16);    // workgroup_mask=0, data_size=2 (4B)
    g1[1] = (int)(256u << 16);  // tensor_dim0 = 256 (low 16 bits at [63:48])
    g1[2] = (int)(1u << 16);    // tensor_dim0 hi=0, tensor_dim1 = 1
    g1[3] = (int)(256u << 16);  // tensor_dim1 hi=0, tile_dim0 = 256
    g1[4] = 1;                  // tile_dim1 = 1, tile_dim2 = 0
    g1[5] = 256;                // tensor_dim0_stride = 256 (low 32)
    g1[6] = (int)(256u << 16);  // stride0 hi=0, tensor_dim1_stride low16 = 256
    g1[7] = 0;
    v4i g2 = {}; v4i g3 = {};
    v8i g4 = {};                // 6-arg (clang-23) form: extra zero group
    __builtin_amdgcn_tensor_load_to_lds(g0, g1, g2, g3, g4, 0);
    (void)lane;
#else
    for (int i = lane; i < EE; i += 32) lds_dst[i] = gsrc[i];
#endif
}
__device__ __forceinline__ void stage_wait() {
#if HAVE_TDM
    __builtin_amdgcn_s_wait_tensorcnt(0);
#endif
}

// ---------------- prep: gather wd/wi, WMMA-based prefix sum S = cumsum(wi) ---
__global__ void __launch_bounds__(1024) prep_kernel(
        const float* __restrict__ W, const int* __restrict__ ar,
        const int* __restrict__ en, float* __restrict__ wd,
        float* __restrict__ wi, float* __restrict__ S)
{
    __shared__ float pref[VV];
    __shared__ float segoff[64];
    const int tid = threadIdx.x;
    const int lane = tid & 31, wave = tid >> 5;

    wd[tid] = W[ar[tid]];
    wi[tid] = W[AA + en[tid]];

    // 64 segments of 16; waves 0..3 each produce 16 segment prefixes via WMMA:
    // D[m,n] = sum_{k<=m,k<16} B[k,n],  B[k,n] = y[16n+k]  (y[0]=0, y[j]=wi[j]).
    // f16 hi/lo split of y accumulated into the f32 C operand -> ~f32 exact.
    if (wave < 4) {
        v16h bhi = {}, blo = {}, amat = {};
        if (lane < 16) { // lanes 0-15 carry B rows K=0..15 (col = lane); K>=16 rows stay 0
            const int seg = (wave << 4) + lane;
#pragma unroll
            for (int k = 0; k < 16; ++k) {
                const int idx = (seg << 4) + k;
                const float y = (idx == 0) ? 0.0f : W[AA + en[idx]];
                const _Float16 h = (_Float16)y;
                bhi[k] = h;
                blo[k] = (_Float16)(y - (float)h);
            }
        }
        const int m = lane & 15;
#pragma unroll
        for (int i = 0; i < 16; ++i) { // lower-triangular ones in A (16x32 layout)
            const int k = ((lane < 16) ? 0 : 16) + i;
            amat[i] = (k < 16 && k <= m) ? (_Float16)1.0f : (_Float16)0.0f;
        }
        v8f acc = {};
        acc = __builtin_amdgcn_wmma_f32_16x16x32_f16(false, amat, false, blo, (short)0, acc, false, false);
        acc = __builtin_amdgcn_wmma_f32_16x16x32_f16(false, amat, false, bhi, (short)0, acc, false, false);
        const int seg   = (wave << 4) + (lane & 15);
        const int pbase = (lane < 16) ? 0 : 8;
#pragma unroll
        for (int r = 0; r < 8; ++r) pref[(seg << 4) + pbase + r] = acc[r];
    }
    __syncthreads();
    if (tid == 0) { // tiny cross-segment carry
        float run = 0.0f;
        for (int s = 0; s < 64; ++s) { segoff[s] = run; run += pref[(s << 4) + 15]; }
    }
    __syncthreads();
    S[tid] = pref[tid] + segoff[tid >> 4];
}

// ---------------- forward/backward lattice DP: block 0 = alpha, block 1 = beta
__global__ void __launch_bounds__(1024) fb_kernel(
        const float* __restrict__ W, const int* __restrict__ ar,
        const int* __restrict__ en, const float* __restrict__ wd,
        const float* __restrict__ S, float* __restrict__ alpha,
        float* __restrict__ beta)
{
    __shared__ float rowbuf[2][VV];
    __shared__ float wsrow[2][EE];
    __shared__ float partial[32];
    __shared__ float scanned[32];

    const int j = threadIdx.x;
    const int lane = j & 31, wave = j >> 5;
    const float Sj = S[j];
    const float* Wsub = W + AA + EE;

    if (blockIdx.x == 0) { // ---- alpha ----
        const int enj = en[j];
        rowbuf[0][j] = Sj;
        alpha[j] = Sj;
        if (wave == 0) stage_row(Wsub + ((long)ar[1] << 8), &wsrow[1][0], lane);
        for (int t = 1; t < TT; ++t) {
            const int cur = t & 1, prv = cur ^ 1;
            if (wave == 0) stage_wait();
            __syncthreads();
            const float wdt  = wd[t];
            const float prevj = rowbuf[prv][j];
            const float diag  = (j > 0) ? (wsrow[cur][enj] + rowbuf[prv][j - 1]) : NEGINF;
            float u = lse2(wdt + prevj, diag) - Sj;      // c[j] - S[j]
            float F = wave_incl(u, lane);
            if (lane == 31) partial[wave] = F;
            __syncthreads();
            if (wave == 0) {
                if (t + 1 < TT) stage_row(Wsub + ((long)ar[t + 1] << 8), &wsrow[prv][0], lane);
                scanned[lane] = wave_incl(partial[lane], lane);
            }
            __syncthreads();
            if (wave > 0) F = lse2(F, scanned[wave - 1]);
            const float av = Sj + F;
            rowbuf[cur][j] = av;
            alpha[(long long)t * VV + j] = av;
        }
    } else { // ---- beta ----
        const float Stot = S[VV - 1];
        const int enj1 = (j < VV - 1) ? en[j + 1] : 0;
        const float binit = Stot - Sj;
        rowbuf[1][j] = binit;
        beta[(long long)(TT - 1) * VV + j] = binit;
        if (wave == 0) stage_row(Wsub + ((long)ar[TT - 1] << 8), &wsrow[0][0], lane);
        for (int t = TT - 2; t >= 0; --t) {
            const int cur = t & 1, prv = cur ^ 1;
            if (wave == 0) stage_wait();
            __syncthreads();
            const float wdt1 = wd[t + 1];
            const float nxtj = rowbuf[prv][j];
            const float diag = (j < VV - 1) ? (wsrow[cur][enj1] + rowbuf[prv][j + 1]) : NEGINF;
            float u = lse2(wdt1 + nxtj, diag) + Sj;      // c[j] + S[j]
            float R = wave_suff(u, lane);
            if (lane == 0) partial[wave] = R;
            __syncthreads();
            if (wave == 0) {
                if (t > 0) stage_row(Wsub + ((long)ar[t] << 8), &wsrow[prv][0], lane);
                scanned[lane] = wave_suff(partial[lane], lane);
            }
            __syncthreads();
            if (wave < 31) R = lse2(R, scanned[wave + 1]);
            const float bv = R - Sj;
            rowbuf[cur][j] = bv;
            beta[(long long)t * VV + j] = bv;
        }
    }
}

// ---------------- scatter-logsumexp: init, max pass, sum pass, finalize ------
__global__ void init_kernel(unsigned int* __restrict__ mkey, float* __restrict__ sacc) {
    int i = blockIdx.x * blockDim.x + threadIdx.x;
    if (i < CC) { mkey[i] = 0u; sacc[i] = 0.0f; }
}

#define N_D ((TT - 1) * VV)
#define N_I (TT * (VV - 1))
#define N_S ((TT - 1) * (VV - 1))
#define N_TOT (N_D + N_I + N_S)

__global__ void countpass(const float* __restrict__ W, const int* __restrict__ ar,
                          const int* __restrict__ en, const float* __restrict__ wd,
                          const float* __restrict__ wi,
                          const float* __restrict__ alpha, const float* __restrict__ beta,
                          unsigned int* __restrict__ mkey, float* __restrict__ sacc,
                          int mode)
{
    long long idx = (long long)blockIdx.x * blockDim.x + threadIdx.x;
    if (idx >= N_TOT) return;
    float val; int id;
    if (idx < N_D) {                                   // deletions
        int t = (int)(idx / VV) + 1, jj = (int)(idx % VV);
        val = alpha[(long long)(t - 1) * VV + jj] + wd[t] + beta[(long long)t * VV + jj];
        id = ar[t];
    } else if (idx < (long long)N_D + N_I) {           // insertions
        long long k = idx - N_D;
        int t = (int)(k / (VV - 1)), jj = (int)(k % (VV - 1)) + 1;
        val = alpha[(long long)t * VV + jj - 1] + wi[jj] + beta[(long long)t * VV + jj];
        id = AA + en[jj];
    } else {                                           // substitutions
        long long k = idx - N_D - N_I;
        int t = (int)(k / (VV - 1)) + 1, jj = (int)(k % (VV - 1)) + 1;
        id = AA + EE + (ar[t] << 8) + en[jj];
        val = alpha[(long long)(t - 1) * VV + jj - 1] + W[id] + beta[(long long)t * VV + jj];
    }
    if (mode == 0) {
        atomicMax(&mkey[id], fkey(val));
    } else {
        float m = fdec(mkey[id]);
        atomicAdd(&sacc[id], expf(val - m));
    }
}

__global__ void finalize(const unsigned int* __restrict__ mkey,
                         const float* __restrict__ sacc, float* __restrict__ out) {
    int i = blockIdx.x * blockDim.x + threadIdx.x;
    if (i >= CC) return;
    unsigned int k = mkey[i];
    out[i] = (k == 0u) ? NEGINF : (fdec(k) + logf(sacc[i]));
}

// ---------------- host entry -------------------------------------------------
extern "C" void kernel_launch(void* const* d_in, const int* in_sizes, int n_in,
                              void* d_out, int out_size, void* d_ws, size_t ws_size,
                              hipStream_t stream) {
    (void)in_sizes; (void)n_in; (void)out_size; (void)ws_size;
    const float* W  = (const float*)d_in[0];
    const int*   ar = (const int*)d_in[1];
    const int*   en = (const int*)d_in[2];
    float* out = (float*)d_out;

    float* wsf   = (float*)d_ws;
    float* alpha = wsf;                       // 1M floats
    float* beta  = alpha + (size_t)TT * VV;   // 1M floats
    float* wd    = beta + (size_t)TT * VV;    // 1K
    float* wi    = wd + TT;                   // 1K
    float* S     = wi + VV;                   // 1K
    unsigned int* mkey = (unsigned int*)(S + VV); // 66048
    float* sacc  = (float*)(mkey + CC);           // 66048

    prep_kernel<<<1, 1024, 0, stream>>>(W, ar, en, wd, wi, S);
    init_kernel<<<(CC + 255) / 256, 256, 0, stream>>>(mkey, sacc);
    fb_kernel<<<2, 1024, 0, stream>>>(W, ar, en, wd, S, alpha, beta);
    int nblk = (int)((N_TOT + 255LL) / 256LL);
    countpass<<<nblk, 256, 0, stream>>>(W, ar, en, wd, wi, alpha, beta, mkey, sacc, 0);
    countpass<<<nblk, 256, 0, stream>>>(W, ar, en, wd, wi, alpha, beta, mkey, sacc, 1);
    finalize<<<(CC + 255) / 256, 256, 0, stream>>>(mkey, sacc, out);
}
